// SpatioTemporalFCAttention_48782238548059
// MI455X (gfx1250) — matrix operand order repaired and checked by reading, not compiled, not run on hardware
//
#include <hip/hip_runtime.h>
#include <hip/hip_bf16.h>

typedef __attribute__((ext_vector_type(16))) _Float16 v16h;
typedef __attribute__((ext_vector_type(8)))  _Float16 v8h;
typedef __attribute__((ext_vector_type(4)))  _Float16 v4h;
typedef __attribute__((ext_vector_type(8)))  float    v8f;

#define B_    512
#define T_    2048
#define D_    256
#define MT    32      // rows of x per workgroup (2048 % 32 == 0 -> never crosses a batch)
#define PITCH 264     // f16 LDS row pitch in halves (+8 pad breaks bank conflicts)

// ---------------------------------------------------------------------------
// A fragment (16x32 f16) from an LDS tile, contiguous k order, same as load_bfrag:
//   lanes 0-15 : M=lane,   halves 0-15 = K[kk*32 + 0..15]
//   lanes 16-31: M=lane-16, halves 0-15 = K[kk*32 + 16..31]
// ---------------------------------------------------------------------------
__device__ __forceinline__ v16h load_afrag(const _Float16* base, int lane, int kk) {
    const int m   = lane & 15;
    const int klo = (lane < 16) ? 0 : 16;
    const _Float16* p = base + m * PITCH + kk * 32 + klo;
    v8h lo = *(const v8h*)(p);
    v8h hi = *(const v8h*)(p + 8);
    return __builtin_shufflevector(lo, hi, 0,1,2,3,4,5,6,7,8,9,10,11,12,13,14,15);
}

// ---------------------------------------------------------------------------
// B fragment (32x16 f16) for y = x @ W^T with row-major W (so B[k][n] = W[n][k]):
//   lane holds column n = ncol + (lane&15); halves j = W[n][kk*32 + kb + j],
//   kb = 0 for lanes 0-15, 16 for lanes 16-31 -> one contiguous 32B load.
// ---------------------------------------------------------------------------
__device__ __forceinline__ v16h load_bfrag(const _Float16* __restrict__ w, int ncol, int lane, int kk) {
    const int n  = ncol + (lane & 15);
    const int kb = kk * 32 + ((lane < 16) ? 0 : 16);
    return *(const v16h*)(w + (size_t)n * D_ + kb);
}

// ---------------------------------------------------------------------------
// Prep: Wi/Wa fp32 -> f16 into workspace; zero the fp32 attention accumulator.
// ---------------------------------------------------------------------------
__global__ __launch_bounds__(256) void prep_kernel(const float* __restrict__ Wi,
                                                   const float* __restrict__ Wa,
                                                   _Float16* __restrict__ wi16,
                                                   _Float16* __restrict__ wa16,
                                                   float* __restrict__ attn_out) {
    const int gid = blockIdx.x * 256 + threadIdx.x;
    if (gid < D_ * D_) {
        wi16[gid] = (_Float16)Wi[gid];
        wa16[gid] = (_Float16)Wa[gid];
    }
    if (gid < B_ * D_) attn_out[gid] = 0.0f;
}

// ---------------------------------------------------------------------------
// Fused: input_o = x@Wi^T + bi + bh ; attn = sigmoid(input_o@Wa^T + ba)
//        attn_out[b,:] += sum_rows(attn * x)   (partial over this WG's 32 rows)
// 8 waves; wave w: mtile = w&1 (16 rows), 4 ntiles starting at (w>>1)*64.
// ---------------------------------------------------------------------------
__global__ __launch_bounds__(256) void fused_attn_kernel(const float* __restrict__ x,
                                                         const _Float16* __restrict__ wi16,
                                                         const _Float16* __restrict__ wa16,
                                                         const float* __restrict__ bi,
                                                         const float* __restrict__ bh,
                                                         const float* __restrict__ ba,
                                                         float* __restrict__ attn_out) {
    __shared__ _Float16 xs[MT][PITCH];
    __shared__ _Float16 ys[MT][PITCH];
    __shared__ float    part[D_];

    const int tid  = threadIdx.x;
    const int lane = tid & 31;
    const int wv   = tid >> 5;
    const int R0   = blockIdx.x * MT;          // global row base in (B*T)
    const int b    = R0 / T_;

    part[tid & (D_ - 1)] = 0.0f;

    // Stage x tile (32 rows x 256 f32 -> f16 LDS). Thread: row=tid/8, 32-col segment.
    {
        const int row = tid >> 3;
        const int seg = tid & 7;
        const float4* xr = (const float4*)(x + (size_t)(R0 + row) * D_) + seg * 8;
        _Float16* dst = &xs[row][seg * 32];
        #pragma unroll
        for (int j = 0; j < 8; ++j) {
            float4 v = xr[j];
            v4h h;
            h[0] = (_Float16)v.x; h[1] = (_Float16)v.y;
            h[2] = (_Float16)v.z; h[3] = (_Float16)v.w;
            *(v4h*)(dst + j * 4) = h;
        }
    }
    __syncthreads();

    const int mbase = (wv & 1) * 16;
    const int ng0   = (wv >> 1) * 64;          // this wave's first output column
    const int moff  = (lane < 16) ? 0 : 8;     // C/D layout: hi lanes hold M+8

    // ---- GEMM1: y1 = x @ Wi^T -------------------------------------------
    v8f acc[4] = {};
    #pragma unroll
    for (int kk = 0; kk < 8; ++kk) {
        v16h a = load_afrag(&xs[mbase][0], lane, kk);
        #pragma unroll
        for (int i = 0; i < 4; ++i) {
            v16h bf = load_bfrag(wi16, ng0 + i * 16, lane, kk);
            acc[i] = __builtin_amdgcn_wmma_f32_16x16x32_f16(
                false, a, false, bf, (short)0, acc[i], false, false);
        }
    }
    // bias (bi + bh, since h0==0 makes hidden_o == bh) and spill to LDS as f16
    #pragma unroll
    for (int i = 0; i < 4; ++i) {
        const int n  = ng0 + i * 16 + (lane & 15);
        const float bb = bi[n] + bh[n];
        #pragma unroll
        for (int r = 0; r < 8; ++r) {
            ys[mbase + moff + r][n] = (_Float16)(acc[i][r] + bb);
        }
    }
    __syncthreads();

    // ---- GEMM2: a = sigmoid(y1 @ Wa^T + ba); reduce a*x over rows --------
    v8f acc2[4] = {};
    #pragma unroll
    for (int kk = 0; kk < 8; ++kk) {
        v16h a = load_afrag(&ys[mbase][0], lane, kk);
        #pragma unroll
        for (int i = 0; i < 4; ++i) {
            v16h bf = load_bfrag(wa16, ng0 + i * 16, lane, kk);
            acc2[i] = __builtin_amdgcn_wmma_f32_16x16x32_f16(
                false, a, false, bf, (short)0, acc2[i], false, false);
        }
    }
    #pragma unroll
    for (int i = 0; i < 4; ++i) {
        const int n  = ng0 + i * 16 + (lane & 15);
        const float bb = ba[n];
        float s = 0.0f;
        #pragma unroll
        for (int r = 0; r < 8; ++r) {
            const float av = 1.0f / (1.0f + __expf(-(acc2[i][r] + bb)));
            s += av * (float)xs[mbase + moff + r][n];
        }
        atomicAdd(&part[n], s);               // ds_add_f32
    }
    __syncthreads();

    atomicAdd(&attn_out[(size_t)b * D_ + tid], part[tid]);  // global_atomic_add_f32
}

// ---------------------------------------------------------------------------
// LSTM cell on a length-1 sequence, h0=c0=0:
//   gates = attn_out @ W_ih^T + b_ih + b_hh ; c = sig(i)*tanh(g) ; h = sig(o)*tanh(c)
// Full fp32 (268 MFLOP, negligible). Block = one batch row.
// ---------------------------------------------------------------------------
__global__ __launch_bounds__(256) void lstm_kernel(const float* __restrict__ attn,
                                                   const float* __restrict__ W_ih,
                                                   const float* __restrict__ b_ih,
                                                   const float* __restrict__ b_hh,
                                                   float* __restrict__ out) {
    __shared__ float arow[D_];
    const int b = blockIdx.x;
    const int h = threadIdx.x;
    arow[h] = attn[(size_t)b * D_ + h];
    __syncthreads();

    float g[4];
    #pragma unroll
    for (int gi = 0; gi < 4; ++gi) {
        const float4* wr = (const float4*)(W_ih + (size_t)(gi * D_ + h) * D_);
        const float4* ar = (const float4*)arow;
        float sx = 0.f, sy = 0.f, sz = 0.f, sw = 0.f;
        #pragma unroll 4
        for (int k = 0; k < D_ / 4; ++k) {
            float4 w4 = wr[k];
            float4 a4 = ar[k];
            sx += w4.x * a4.x; sy += w4.y * a4.y;
            sz += w4.z * a4.z; sw += w4.w * a4.w;
        }
        g[gi] = b_ih[gi * D_ + h] + b_hh[gi * D_ + h] + ((sx + sy) + (sz + sw));
    }
    const float ig = 1.0f / (1.0f + __expf(-g[0]));
    const float og = 1.0f / (1.0f + __expf(-g[3]));
    const float gg = tanhf(g[2]);
    const float c  = ig * gg;                  // f*c0 == 0
    const float hh = og * tanhf(c);

    const int idx = b * D_ + h;
    const int BH  = B_ * D_;
    out[idx]          = hh;   // lstm_output
    out[BH + idx]     = hh;   // h_n
    out[2 * BH + idx] = c;    // c_n
}

// ---------------------------------------------------------------------------
extern "C" void kernel_launch(void* const* d_in, const int* in_sizes, int n_in,
                              void* d_out, int out_size, void* d_ws, size_t ws_size,
                              hipStream_t stream) {
    (void)in_sizes; (void)n_in; (void)out_size; (void)ws_size;
    // setup_inputs() dict order:
    const float* x    = (const float*)d_in[0];
    const float* Wi   = (const float*)d_in[1];
    const float* bi   = (const float*)d_in[2];
    // d_in[3] = Wh   (unused: h0 == 0)
    const float* bh   = (const float*)d_in[4];
    const float* Wa   = (const float*)d_in[5];
    const float* ba   = (const float*)d_in[6];
    const float* W_ih = (const float*)d_in[7];
    const float* b_ih = (const float*)d_in[8];
    // d_in[9] = W_hh (unused: h0 == 0)
    const float* b_hh = (const float*)d_in[10];
    float* out = (float*)d_out;

    char* ws = (char*)d_ws;
    float*    attn_out = (float*)ws;                              // 512 KB
    _Float16* wi16     = (_Float16*)(ws + 512 * 1024);            // 128 KB
    _Float16* wa16     = (_Float16*)(ws + 512 * 1024 + 128 * 1024); // 128 KB

    prep_kernel<<<(B_ * D_) / 256, 256, 0, stream>>>(Wi, Wa, wi16, wa16, attn_out);
    fused_attn_kernel<<<(B_ * T_) / MT, 256, 0, stream>>>(x, wi16, wa16, bi, bh, ba, attn_out);
    lstm_kernel<<<B_, 256, 0, stream>>>(attn_out, W_ih, b_ih, b_hh, out);
}
